// GAT_29162827939936
// MI455X (gfx1250) — matrix operand order repaired and checked by reading, not compiled, not run on hardware
//
#include <hip/hip_runtime.h>
#include <hip/hip_bf16.h>

#define GN 100000
#define GE 1600000
#define GIN 128
#define GH 4

typedef _Float16 f16;
typedef __attribute__((ext_vector_type(16))) _Float16 v16h;
typedef __attribute__((ext_vector_type(8)))  _Float16 v8h;
typedef __attribute__((ext_vector_type(8)))  float    v8f;

__device__ __forceinline__ void atomic_add_f32(float* p, float v) {
    __hip_atomic_fetch_add(p, v, __ATOMIC_RELAXED, __HIP_MEMORY_SCOPE_AGENT);
}

// ---------------- conversion kernels ----------------
__global__ void cvt_f32_to_f16(const float* __restrict__ in, f16* __restrict__ out, int n) {
    int i = blockIdx.x * blockDim.x + threadIdx.x;
    if (i < n) out[i] = (f16)in[i];
}

// Pack weights straight into WMMA B-fragment order.
// Logical B is [K x NCOLS]: cols [0,split) from W0 (row stride split),
// cols [split,NCOLS) from W1opt (row stride NCOLS-split).
// Output layout: frag[ct][ks][lane][0..15], 16 contiguous halves per lane:
//   elem 2v   -> B[ks*32 + (lane>>4)*16 + 2v    ][ct*16 + (lane&15)]
//   elem 2v+1 -> B[ks*32 + (lane>>4)*16 + 2v + 1][ct*16 + (lane&15)]
__global__ void pack_b_frag(const float* __restrict__ W0, const float* __restrict__ W1opt,
                            f16* __restrict__ bsw, int K, int ncols, int split) {
    int i = blockIdx.x * blockDim.x + threadIdx.x;
    int nk = K >> 5;
    int total = (ncols >> 4) * nk * 32;
    if (i >= total) return;
    int lane = i & 31;
    int t    = i >> 5;
    int ks   = t % nk;
    int ct   = t / nk;
    int half = lane >> 4, l15 = lane & 15;
    int col  = ct * 16 + l15;
    f16* outp = bsw + i * 16;
#pragma unroll
    for (int v = 0; v < 8; ++v) {
        int k = ks * 32 + half * 16 + 2 * v;
        float e0, e1;
        if (col < split) {
            e0 = W0[k * split + col];
            e1 = W0[(k + 1) * split + col];
        } else {
            int w = ncols - split, c = col - split;
            e0 = W1opt[k * w + c];
            e1 = W1opt[(k + 1) * w + c];
        }
        outp[2 * v]     = (f16)e0;
        outp[2 * v + 1] = (f16)e1;
    }
}

// ---------------- WMMA GEMM ----------------
// C = A[Nrows x K] @ B[K x NCOLS], A row-major f16, B pre-swizzled fragments.
// Block: ROWT row-tiles x (NCOLS/16) col-tiles, one wave each.
// Col tiles [0,SPLIT) -> C0, [SPLIT,NCOLS) -> C1; both have row stride 64.
template <int K, int NCOLS, int ROWT, int SPLIT>
__global__ void gemm_wmma_f16(const f16* __restrict__ A, const f16* __restrict__ Bsw,
                              float* __restrict__ C0, float* __restrict__ C1) {
    constexpr int CT = NCOLS / 16;   // col tiles per block
    constexpr int NK = K / 32;       // K steps
    const int wave = threadIdx.x >> 5;
    const int lane = threadIdx.x & 31;
    const int half = lane >> 4;
    const int l15  = lane & 15;
    const int ct   = wave % CT;
    const int rt   = wave / CT;
    const int row0 = (blockIdx.x * ROWT + rt) * 16;

    v8f c = {};
    const f16* Arow = A + (row0 + l15) * K;
    const f16* Bp   = Bsw + (ct * NK * 32 + lane) * 16;

#pragma unroll
    for (int ks = 0; ks < NK; ++ks) {
        v16h a = *(const v16h*)(Arow + ks * 32 + half * 16);       // K = ks*32+h*16 .. +15
        v16h b = *(const v16h*)(Bp + ks * 32 * 16);
        c = __builtin_amdgcn_wmma_f32_16x16x32_f16(false, a, false, b, (short)0, c,
                                                   false, false);
    }

    // C/D layout: VGPR r, lanes 0-15 -> M=r, lanes 16-31 -> M=r+8; N = lane&15.
    // Wave-uniform output selection (SPLIT is a multiple of 16); both outputs stride 64.
    const int colB = ct * 16;
    float* outp = (colB < SPLIT) ? (C0 + colB)
                                 : (C1 + (colB - SPLIT));
    outp += (row0 + half * 8) * 64 + l15;
#pragma unroll
    for (int r = 0; r < 8; ++r)
        outp[r * 64] = c[r];
}

// ---------------- attention coefficients ----------------
__global__ void node_attn(const float* __restrict__ feat, const float* __restrict__ al,
                          const float* __restrict__ ar, float* __restrict__ el,
                          float* __restrict__ er, int NH) {
    int i = blockIdx.x * blockDim.x + threadIdx.x;
    if (i >= NH) return;
    int h = i & (GH - 1);
    const float* f = feat + (i >> 2) * 64 + h * 16;
    const float* a = al + h * 16;
    const float* b = ar + h * 16;
    float sl = 0.f, sr = 0.f;
#pragma unroll
    for (int d = 0; d < 16; ++d) { float v = f[d]; sl += v * a[d]; sr += v * b[d]; }
    el[i] = sl;
    er[i] = sr;
}

// ---------------- edge pass A: softmax denominators ----------------
__global__ void edge_den(const int* __restrict__ src, const int* __restrict__ dst,
                         const float* __restrict__ el, const float* __restrict__ er,
                         float* __restrict__ den, int EH) {
    int i = blockIdx.x * blockDim.x + threadIdx.x;
    if (i >= EH) return;
    int e = i >> 2, h = i & (GH - 1);
    int s = src[e], d = dst[e];
    float v = el[s * GH + h] + er[d * GH + h];
    v = v > 0.f ? v : 0.2f * v;            // leaky_relu
    atomic_add_f32(&den[d * GH + h], expf(v));
}

// ---------------- edge pass B: weighted message aggregation ----------------
__global__ void edge_agg(const int* __restrict__ src, const int* __restrict__ dst,
                         const float* __restrict__ el, const float* __restrict__ er,
                         const float* __restrict__ den, const float* __restrict__ feat,
                         float* __restrict__ acc, int EH) {
    int i = blockIdx.x * blockDim.x + threadIdx.x;
    if (i >= EH) return;
    int e = i >> 2, h = i & (GH - 1);
    int s = src[e], d = dst[e];
    float v = el[s * GH + h] + er[d * GH + h];
    v = v > 0.f ? v : 0.2f * v;
    float alpha = expf(v) / den[d * GH + h];
    const float4* fs = (const float4*)(feat + s * 64 + h * 16);
    float* ad = acc + d * 64 + h * 16;
#pragma unroll
    for (int q = 0; q < 4; ++q) {
        float4 f = fs[q];
        atomic_add_f32(ad + 4 * q + 0, alpha * f.x);
        atomic_add_f32(ad + 4 * q + 1, alpha * f.y);
        atomic_add_f32(ad + 4 * q + 2, alpha * f.z);
        atomic_add_f32(ad + 4 * q + 3, alpha * f.w);
    }
}

// ---------------- layer-1 epilogue ----------------
__global__ void finalize1(const float* __restrict__ acc, float* __restrict__ res_h1,
                          const float* __restrict__ b1, f16* __restrict__ h1h, int n) {
    int i = blockIdx.x * blockDim.x + threadIdx.x;
    if (i >= n) return;
    int c = i & 63;
    float v = acc[i] + res_h1[i] + b1[c];
    v = v > 0.f ? v : expm1f(v);           // elu
    res_h1[i] = v;                         // becomes h1 (residual for layer 2)
    h1h[i] = (f16)v;
}

// ---------------- layer-2 epilogue: +res +bias, head mean ----------------
__global__ void finalize2(const float* __restrict__ acc, const float* __restrict__ h1,
                          const float* __restrict__ b2, float* __restrict__ out, int n) {
    int i = blockIdx.x * blockDim.x + threadIdx.x;
    if (i >= n) return;
    int nd = i >> 4, d = i & 15;
    float s = 0.f;
#pragma unroll
    for (int h = 0; h < GH; ++h) {
        int c = h * 16 + d;
        s += acc[nd * 64 + c] + h1[nd * 64 + c] + b2[c];
    }
    out[i] = s * 0.25f;
}

extern "C" void kernel_launch(void* const* d_in, const int* in_sizes, int n_in,
                              void* d_out, int out_size, void* d_ws, size_t ws_size,
                              hipStream_t stream) {
    const float* x     = (const float*)d_in[0];
    const int*   src   = (const int*)d_in[1];
    const int*   dst   = (const int*)d_in[2];
    const float* W1    = (const float*)d_in[3];
    const float* al1   = (const float*)d_in[4];
    const float* ar1   = (const float*)d_in[5];
    const float* resW1 = (const float*)d_in[6];
    const float* b1    = (const float*)d_in[7];
    const float* W2    = (const float*)d_in[8];
    const float* al2   = (const float*)d_in[9];
    const float* ar2   = (const float*)d_in[10];
    const float* b2    = (const float*)d_in[11];
    float* out = (float*)d_out;

    // workspace carve-up (256B aligned)
    char*  ws  = (char*)d_ws;
    size_t off = 0;
    auto carve = [&](size_t bytes) -> char* {
        char* p = ws + off;
        off += (bytes + 255) & ~(size_t)255;
        return p;
    };
    f16*   xh    = (f16*)carve((size_t)GN * GIN * 2);   // x in f16
    f16*   bsw1  = (f16*)carve(128 * 128 * 2);          // [W1|resW1] swizzled frags
    f16*   bsw2  = (f16*)carve(64 * 64 * 2);            // W2 swizzled frags
    float* feat  = (float*)carve((size_t)GN * 64 * 4);  // layer feat (reused)
    float* res1  = (float*)carve((size_t)GN * 64 * 4);  // res1 -> h1 in place
    float* el    = (float*)carve((size_t)GN * GH * 4);
    float* er    = (float*)carve((size_t)GN * GH * 4);
    float* den   = (float*)carve((size_t)GN * GH * 4);
    float* acc   = (float*)carve((size_t)GN * 64 * 4);
    f16*   h1h   = (f16*)carve((size_t)GN * 64 * 2);    // h1 f16 for layer-2 GEMM

    const int TB = 256;
    const int NH = GN * GH;                 // 400000
    const int EH = GE * GH;                 // 6.4M

    // --- conversions / weight packing ---
    cvt_f32_to_f16<<<(GN * GIN + TB - 1) / TB, TB, 0, stream>>>(x, xh, GN * GIN);
    pack_b_frag<<<(8 * 4 * 32 + TB - 1) / TB, TB, 0, stream>>>(W1, resW1, bsw1, 128, 128, 64);
    pack_b_frag<<<(4 * 2 * 32 + TB - 1) / TB, TB, 0, stream>>>(W2, W2, bsw2, 64, 64, 64);

    // --- layer 1: [feat1 | res1] = x @ [W1 | resW1] ---
    gemm_wmma_f16<128, 128, 1, 64><<<GN / 16, 256, 0, stream>>>(xh, bsw1, feat, res1);
    node_attn<<<(NH + TB - 1) / TB, TB, 0, stream>>>(feat, al1, ar1, el, er, NH);
    hipMemsetAsync(den, 0, (size_t)GN * GH * 4, stream);
    hipMemsetAsync(acc, 0, (size_t)GN * 64 * 4, stream);
    edge_den<<<(EH + TB - 1) / TB, TB, 0, stream>>>(src, dst, el, er, den, EH);
    edge_agg<<<(EH + TB - 1) / TB, TB, 0, stream>>>(src, dst, el, er, den, feat, acc, EH);
    finalize1<<<(GN * 64 + TB - 1) / TB, TB, 0, stream>>>(acc, res1, b1, h1h, GN * 64);

    // --- layer 2 (res1 holds h1; identity residual) ---
    gemm_wmma_f16<64, 64, 2, 64><<<GN / 32, 256, 0, stream>>>(h1h, bsw2, feat, feat);
    node_attn<<<(NH + TB - 1) / TB, TB, 0, stream>>>(feat, al2, ar2, el, er, NH);
    hipMemsetAsync(den, 0, (size_t)GN * GH * 4, stream);
    hipMemsetAsync(acc, 0, (size_t)GN * 64 * 4, stream);
    edge_den<<<(EH + TB - 1) / TB, TB, 0, stream>>>(src, dst, el, er, den, EH);
    edge_agg<<<(EH + TB - 1) / TB, TB, 0, stream>>>(src, dst, el, er, den, feat, acc, EH);
    finalize2<<<(GN * 16 + TB - 1) / TB, TB, 0, stream>>>(acc, res1, b2, out, GN * 16);
}
